// precond_net_20572893347856
// MI455X (gfx1250) — matrix operand order repaired and checked by reading, not compiled, not run on hardware
//
#include <hip/hip_runtime.h>
#include <hip/hip_bf16.h>

// MI455X / gfx1250, wave32. Implicit-GEMM convolution on v_wmma_f32_16x16x32_bf16.
// Each wave owns a 32(m)x32(n) register tile = 4 independent accumulators.
// SAME-padding handled via a zero page: OOB lanes read zeros through a
// redirected base pointer (2 cndmasks per tap, hoisted out of the K loop).

typedef __attribute__((ext_vector_type(16))) __bf16 v16bf;
typedef __attribute__((ext_vector_type(8)))  __bf16 v8bf;
typedef __attribute__((ext_vector_type(8)))  float  v8f;

#define HH 64
#define WWID 64
#define BBATCH 4
#define NPIX (BBATCH * HH * WWID)   // 16384 output pixels

// ---------------------------------------------------------------------------
// Fragment loaders for the CDNA5 16-bit WMMA VGPR layouts.
// A (16x32, MxK): lane<16 -> row M=lane,   K {0..7}  then {16..23}
//                 lane>=16-> row M=lane-16,K {8..15} then {24..31}
//   caller passes p already offset by +8 for the hi half-wave.
__device__ __forceinline__ v16bf ldA(const __bf16* p) {
  v8bf lo = *(const v8bf*)p;
  v8bf hi = *(const v8bf*)(p + 16);
  return __builtin_shufflevector(lo, hi, 0,1,2,3,4,5,6,7,8,9,10,11,12,13,14,15);
}
// B (32x16, KxN): lane<16 -> col N=lane, K {0..7} then {16..23};
//                 lane>=16 -> K {8..15} then {24..31}.
//   caller passes p already offset by +8 for the hi half-wave -> two global_load_b128.
__device__ __forceinline__ v16bf ldB(const __bf16* p) {
  v8bf lo = *(const v8bf*)p;
  v8bf hi = *(const v8bf*)(p + 16);
  return __builtin_shufflevector(lo, hi, 0,1,2,3,4,5,6,7,8,9,10,11,12,13,14,15);
}
__device__ __forceinline__ v8f wmma_bf16(v16bf a, v16bf b, v8f c) {
  return __builtin_amdgcn_wmma_f32_16x16x32_bf16(false, a, false, b,
                                                 (short)0, c, false, false);
}

// ---------------------------------------------------------------------------
// Zero the padding page (re-done every launch: harness poisons the workspace).
// ---------------------------------------------------------------------------
__global__ void k_zero(float* __restrict__ p, int n_dwords) {
  int i = blockIdx.x * 256 + threadIdx.x;
  if (i < n_dwords) p[i] = 0.f;
}

// ---------------------------------------------------------------------------
// Prepack weights: (co, ci, k, k) f32  ->  (co, k*k, ci) bf16
// ---------------------------------------------------------------------------
__global__ void k_prepack(const float* __restrict__ w, __bf16* __restrict__ wp,
                          int co, int ci, int k) {
  long i = (long)blockIdx.x * 256 + threadIdx.x;
  long total = (long)co * ci * k * k;
  if (i >= total) return;
  int  c   = (int)(i % ci);
  long t   = i / ci;
  int  kyx = (int)(t % (k * k));
  int  o   = (int)(t / (k * k));
  wp[i] = (__bf16)w[((long)o * ci + c) * (k * k) + kyx];
}

// ---------------------------------------------------------------------------
// Layer 0: 1 -> 64 channels, 1x1 conv + bias + PReLU, write NHWC bf16
// ---------------------------------------------------------------------------
__global__ void k_layer0(const float* __restrict__ A, const float* __restrict__ w,
                         const float* __restrict__ bias, const float* __restrict__ al,
                         __bf16* __restrict__ out) {
  int i = blockIdx.x * 256 + threadIdx.x;          // over NPIX*64
  if (i >= NPIX * 64) return;
  int c = i & 63;
  int p = i >> 6;
  float v = A[p] * w[c] + bias[c];
  float a = al[0];
  v = (v >= 0.f) ? v : a * v;
  out[(long)p * 64 + c] = (__bf16)v;
}

// ---------------------------------------------------------------------------
// Implicit-GEMM conv (SAME, stride 1). GEMM: M=Cout, N=16384 pixels,
// K = k*k*Cin in chunks of 32. Block = 256 thr = 8 waves, 2(m)x4(n) wave grid
// -> 64x128 block tile; wave tile 32x32 (acc00/01/10/11).
// ---------------------------------------------------------------------------
template <int CIN, int KK, int PAD>
__global__ __launch_bounds__(256) void k_conv_wmma(
    const __bf16* __restrict__ act,   // (B,H,W,CIN) bf16
    const __bf16* __restrict__ wp,    // (Cout, KK*KK, CIN) bf16
    const float*  __restrict__ bias,  // (Cout)
    const float*  __restrict__ alpha, // (1)
    const __bf16* __restrict__ zp,    // zero page (>= CIN zeros)
    __bf16*       __restrict__ out,   // (B,H,W,Cout) bf16
    int Cout) {
  const int lane = threadIdx.x & 31;
  const int wid  = threadIdx.x >> 5;
  const int wm   = wid & 1;                      // 2 m-waves
  const int wn   = wid >> 1;                     // 4 n-waves
  const int m0   = blockIdx.y * 64 + wm * 32;    // wave rows: m0, m0+16
  const int n0   = blockIdx.x * 128 + wn * 32;   // wave cols: n0, n0+16
  const int l    = lane & 15;
  const int hi   = lane >> 4;

  // Wave's 32 pixels are consecutive x in one image row (n0 % 32 == 0, row=64),
  // so (b, y) are wave-uniform; frag1's x is frag0's x + 16.
  const int nA = n0 + l;
  const int x0 = nA & 63;
  const int y  = (nA >> 6) & 63;
  const int b  = nA >> 12;

  constexpr long Krow = (long)KK * KK * CIN;
  const __bf16* a0row = wp + (long)(m0 + l) * Krow + hi * 8;

  v8f acc00 = {}, acc01 = {}, acc10 = {}, acc11 = {};

  for (int ky = 0; ky < KK; ++ky) {
    const int  yy  = y + ky - PAD;
    const bool yok = (yy >= 0) & (yy < HH);
    const long rowbase = (long)((b * HH + (yok ? yy : 0)) * WWID) * CIN;
    // Pull the next activation row toward the WGP while this tap computes.
    if (ky + 1 < KK) {
      const int yn = y + ky + 1 - PAD;
      if (yn >= 0 && yn < HH)
        __builtin_prefetch(act + (long)((b * HH + yn) * WWID + x0) * CIN, 0, 1);
    }
    for (int kx = 0; kx < KK; ++kx) {
      const int  xa = x0 + kx - PAD;
      const int  xb = xa + 16;
      const bool va = yok & (xa >= 0) & (xa < WWID);
      const bool vb = yok & (xb >= 0) & (xb < WWID);
      // Zero-page redirect: OOB lanes stream zeros; no per-element selects.
      const __bf16* browa = (va ? act + rowbase + (long)xa * CIN : zp) + hi * 8;
      const __bf16* browb = (vb ? act + rowbase + (long)xb * CIN : zp) + hi * 8;
      const __bf16* ap0   = a0row + (long)(ky * KK + kx) * CIN;
      const __bf16* ap1   = ap0 + 16 * Krow;
#pragma unroll 4
      for (int cb = 0; cb < CIN; cb += 32) {
        v16bf af0 = ldA(ap0 + cb);
        v16bf af1 = ldA(ap1 + cb);
        v16bf bf0 = ldB(browa + cb);
        v16bf bf1 = ldB(browb + cb);
        acc00 = wmma_bf16(af0, bf0, acc00);
        acc01 = wmma_bf16(af0, bf1, acc01);
        acc10 = wmma_bf16(af1, bf0, acc10);
        acc11 = wmma_bf16(af1, bf1, acc11);
      }
    }
  }

  // Epilogue: C/D layout -> lane holds col N=l, rows M = r + 8*hi. Each lane's
  // 8 values are 8 consecutive output channels in NHWC -> one 16B store each.
  const float a = alpha[0];
  auto emit = [&](v8f acc, int mbase, int n) {
    const int cobase = mbase + hi * 8;
    v8bf o;
#pragma unroll
    for (int r = 0; r < 8; ++r) {
      float v = acc[r] + bias[cobase + r];
      v = (v >= 0.f) ? v : a * v;
      o[r] = (__bf16)v;
    }
    *(v8bf*)(out + (long)n * Cout + cobase) = o;
  };
  emit(acc00, m0,      nA);
  emit(acc01, m0,      nA + 16);
  emit(acc10, m0 + 16, nA);
  emit(acc11, m0 + 16, nA + 16);
}

// ---------------------------------------------------------------------------
// Layer 7: 64 -> 1 channels, 1x1 conv + bias + PReLU; produce m (B,64,64) f32
// ---------------------------------------------------------------------------
__global__ void k_layer7(const __bf16* __restrict__ act, const float* __restrict__ w,
                         const float* __restrict__ bias, const float* __restrict__ al,
                         float* __restrict__ m) {
  int p = blockIdx.x * 256 + threadIdx.x;
  if (p >= NPIX) return;
  const __bf16* ap = act + (long)p * 64;
  float s = bias[0];
#pragma unroll 8
  for (int c = 0; c < 64; ++c) s += (float)ap[c] * w[c];
  float a = al[0];
  m[p] = (s >= 0.f) ? s : a * s;
}

// ---------------------------------------------------------------------------
// Gram: out[b,i,k] = sum_j m[b,i,j]*m[b,k,j]  (already symmetric) — 2 MFLOP
// ---------------------------------------------------------------------------
__global__ void k_gram(const float* __restrict__ m, float* __restrict__ out) {
  int i = blockIdx.x * 256 + threadIdx.x;          // over B*64*64
  if (i >= BBATCH * 64 * 64) return;
  int kk = i & 63;
  int ii = (i >> 6) & 63;
  int b  = i >> 12;
  const float* r1 = m + (b * 64 + ii) * 64;
  const float* r2 = m + (b * 64 + kk) * 64;
  float s = 0.f;
#pragma unroll 8
  for (int j = 0; j < 64; ++j) s += r1[j] * r2[j];
  out[i] = s;
}

// ---------------------------------------------------------------------------
extern "C" void kernel_launch(void* const* d_in, const int* in_sizes, int n_in,
                              void* d_out, int out_size, void* d_ws, size_t ws_size,
                              hipStream_t stream) {
  const float* A = (const float*)d_in[0];
  auto Wf = [&](int i) { return (const float*)d_in[1 + 3 * i]; };
  auto Bf = [&](int i) { return (const float*)d_in[2 + 3 * i]; };
  auto Af = [&](int i) { return (const float*)d_in[3 + 3 * i]; };

  // Workspace layout (~53 MB):
  //   [0, 16MB)        act ping   (max B,H,W,512 bf16)
  //   [16, 32MB)       act pong
  //   [32MB, +64K)     m buffer f32
  //   [32MB+64K, +2K)  zero page
  //   [33MB, ...)      packed weights bf16, layers 1..6
  char* ws = (char*)d_ws;
  __bf16* actP[2] = { (__bf16*)ws, (__bf16*)(ws + (16u << 20)) };
  float*  mbuf    = (float*)(ws + (32u << 20));
  __bf16* zpage   = (__bf16*)(ws + (32u << 20) + (64u << 10));
  char*   wpcur   = ws + (33u << 20);

  struct Cfg { int ci, co, k; };
  const Cfg cfg[6] = {
    { 64, 128, 20}, {128, 256, 3}, {256, 512, 3},
    {512, 256, 3},  {256, 128, 3}, {128, 64, 20},
  };

  // Zero page (2KB) — re-zero every call (workspace is poisoned by harness).
  k_zero<<<2, 256, 0, stream>>>((float*)zpage, 512);

  // Prepack bf16 weights for the 6 WMMA layers (deterministic, every call).
  __bf16* wp[6];
  for (int i = 0; i < 6; ++i) {
    const Cfg c = cfg[i];
    long total = (long)c.co * c.ci * c.k * c.k;
    wp[i] = (__bf16*)wpcur;
    wpcur += total * sizeof(__bf16);
    k_prepack<<<(int)((total + 255) / 256), 256, 0, stream>>>(
        Wf(i + 1), wp[i], c.co, c.ci, c.k);
  }

  // Layer 0: scalar 1x1 expansion to NHWC bf16.
  k_layer0<<<(NPIX * 64) / 256, 256, 0, stream>>>(A, Wf(0), Bf(0), Af(0), actP[0]);

  // Layers 1..6: WMMA implicit GEMM, ping-pong activations.
  // grid = (N/128, Cout/64), block = 256.
#define CONV(idx, CI, KS, PD, CO)                                             \
  k_conv_wmma<CI, KS, PD><<<dim3(NPIX / 128, (CO) / 64), 256, 0, stream>>>(   \
      actP[cur], wp[idx], Bf((idx) + 1), Af((idx) + 1), zpage, actP[cur ^ 1], \
      (CO));                                                                  \
  cur ^= 1;

  int cur = 0;
  CONV(0,  64, 20, 9, 128)
  CONV(1, 128,  3, 1, 256)
  CONV(2, 256,  3, 1, 512)
  CONV(3, 512,  3, 1, 256)
  CONV(4, 256,  3, 1, 128)
  CONV(5, 128, 20, 9,  64)
#undef CONV

  // Layer 7: reduce 64 channels -> m, then Gram -> output (f32, 4x64x64).
  k_layer7<<<NPIX / 256, 256, 0, stream>>>(actP[cur], Wf(7), Bf(7), Af(7), mbuf);
  k_gram<<<(BBATCH * 64 * 64) / 256, 256, 0, stream>>>(mbuf, (float*)d_out);
}
